// DecoderBlock_65498251264371
// MI455X (gfx1250) — matrix operand order repaired and checked by reading, not compiled, not run on hardware
//
#include <hip/hip_runtime.h>

// ---- CDNA5 (gfx1250) wave32 WMMA decoder block -----------------------------
// All GEMMs and attention contractions run through v_wmma_f32_16x16x32_f16.

typedef __attribute__((ext_vector_type(16))) _Float16 v16h;
typedef __attribute__((ext_vector_type(8)))  _Float16 h8;
typedef __attribute__((ext_vector_type(4)))  _Float16 h4;
typedef __attribute__((ext_vector_type(2)))  _Float16 h2;
typedef __attribute__((ext_vector_type(8)))  float    v8f;
typedef __attribute__((ext_vector_type(4)))  float    f4;

#define D_MODEL 1024
#define S_LEN   1024
#define NB      4
#define NH      16
#define DK      64
#define FFN_DIM 4096

__device__ __forceinline__ v8f wmma_f16(v16h a, v16h b, v8f c) {
  return __builtin_amdgcn_wmma_f32_16x16x32_f16(false, a, false, b, (short)0, c,
                                                false, false);
}

union HV { v16h v; h8 h[2]; };

__device__ __forceinline__ v8f zero8() {
  v8f z = {0.f, 0.f, 0.f, 0.f, 0.f, 0.f, 0.f, 0.f};
  return z;
}

// ---------------------------------------------------------------------------
// GEMM: C[M,N] = A[M,K] (f32) * B[K,N] (f32) + bias, f16 compute / f32 accum.
// OutT: float or _Float16.  RELU: fuse relu.  VT: store f16 transposed as
// V^T [b,h,d,s] for attention.  HEADED: B is stacked [H, K, 64] per-head.
// Block: 256 threads (8 waves), 64x64 tile, K-step 32.
// ---------------------------------------------------------------------------
template<typename OutT, bool RELU, bool VT, bool HEADED>
__global__ __launch_bounds__(256) void gemm_wmma(
    const float* __restrict__ A, const float* __restrict__ B,
    const float* __restrict__ bias, void* __restrict__ Cp,
    int M, int N, int K)
{
  __shared__ __align__(16) _Float16 As[64][40];
  __shared__ __align__(16) _Float16 Bs[64][40];

  const int tid  = threadIdx.x;
  const int m0   = blockIdx.y * 64;
  const int n0   = blockIdx.x * 64;
  const int w    = tid >> 5;
  const int lane = tid & 31;
  const int lrow = lane & 15;
  const int half = lane >> 4;
  const int r    = w >> 1;          // wave's 16-row block (0..3)
  const int c    = (w & 1) * 32;    // wave's 32-col base (two 16x16 tiles)

  v8f acc0 = zero8();
  v8f acc1 = zero8();

  for (int k0 = 0; k0 < K; k0 += 32) {
    // Stage A tile 64x32 (f32 -> f16): float4 loads, packed ds_store_b64
    for (int i = 0; i < 2; ++i) {
      int q  = tid + i * 256;            // 0..511 quads
      int mm = q >> 3;                   // 8 quads per 32-wide row
      int kq = (q & 7) * 4;
      f4 a4 = *(const f4*)&A[(size_t)(m0 + mm) * K + (k0 + kq)];
      h4 hv;
      hv[0] = (_Float16)a4[0]; hv[1] = (_Float16)a4[1];
      hv[2] = (_Float16)a4[2]; hv[3] = (_Float16)a4[3];
      *(h4*)&As[mm][kq] = hv;
    }
    // Stage B tile 32x64 transposed -> Bs[n][k]: k-pairs, ds_store_b32
    for (int i = 0; i < 4; ++i) {
      int p  = tid + i * 256;            // 0..1023 pairs
      int kp = (p >> 6) * 2;             // even k
      int nn = p & 63;
      int gn = n0 + nn;
      float b0v, b1v;
      if (HEADED) {
        // stacked per-head weight [H][K][64]: col gn = h*64 + dk
        size_t base = (size_t)(gn >> 6) * (size_t)K * 64 +
                      (size_t)(k0 + kp) * 64 + (gn & 63);
        b0v = B[base];
        b1v = B[base + 64];
      } else {
        size_t base = (size_t)(k0 + kp) * N + gn;
        b0v = B[base];
        b1v = B[base + N];
      }
      h2 hv;
      hv[0] = (_Float16)b0v; hv[1] = (_Float16)b1v;
      *(h2*)&Bs[nn][kp] = hv;
    }
    __syncthreads();

    // A fragment: lanes 0-15 hold K {0..7,16..23}; lanes 16-31 K {8..15,24..31}
    HV a, b0, b1;
    a.h[0] = *(const h8*)&As[r * 16 + lrow][half * 8];
    a.h[1] = *(const h8*)&As[r * 16 + lrow][16 + half * 8];
    // B fragment: lanes 0-15 hold K {0..7,16..23}; lanes 16-31 K {8..15,24..31}
    b0.h[0] = *(const h8*)&Bs[c + lrow][half * 8];
    b0.h[1] = *(const h8*)&Bs[c + lrow][16 + half * 8];
    b1.h[0] = *(const h8*)&Bs[c + 16 + lrow][half * 8];
    b1.h[1] = *(const h8*)&Bs[c + 16 + lrow][16 + half * 8];

    acc0 = wmma_f16(a.v, b0.v, acc0);
    acc1 = wmma_f16(a.v, b1.v, acc1);
    __syncthreads();
  }

  // Epilogue. C layout: VGPR i -> row = i + (half?8:0), col = lrow.
  const int gn0 = n0 + c + lrow;
  const int gn1 = gn0 + 16;
  const float bias0 = bias[gn0];
  const float bias1 = bias[gn1];
  for (int i = 0; i < 8; ++i) {
    int gm = m0 + r * 16 + i + half * 8;
    float v0 = acc0[i] + bias0;
    float v1 = acc1[i] + bias1;
    if (RELU) { v0 = fmaxf(v0, 0.f); v1 = fmaxf(v1, 0.f); }
    if constexpr (VT) {
      // store f16 V^T: [b][h][d][s]
      int bb = gm >> 10, ss = gm & 1023;
      ((_Float16*)Cp)[(((size_t)bb * NH + (gn0 >> 6)) * DK + (gn0 & 63)) *
                          S_LEN + ss] = (_Float16)v0;
      ((_Float16*)Cp)[(((size_t)bb * NH + (gn1 >> 6)) * DK + (gn1 & 63)) *
                          S_LEN + ss] = (_Float16)v1;
    } else if constexpr (sizeof(OutT) == 2) {
      ((_Float16*)Cp)[(size_t)gm * N + gn0] = (_Float16)v0;
      ((_Float16*)Cp)[(size_t)gm * N + gn1] = (_Float16)v1;
    } else {
      ((float*)Cp)[(size_t)gm * N + gn0] = v0;
      ((float*)Cp)[(size_t)gm * N + gn1] = v1;
    }
  }
}

// ---------------------------------------------------------------------------
// Flash attention. q,k: f16 [b,s,h,d]; vT: f16 [b,h,d,s]; out: f32 [b,s,h*d].
// Block = 128 threads (4 waves); each wave owns 16 query rows, streams keys
// in blocks of 32 with online softmax.  scale = 1/sqrt(64) = 0.125.
// ---------------------------------------------------------------------------
template<bool CAUSAL>
__global__ __launch_bounds__(128) void attn_wmma(
    const _Float16* __restrict__ q, const _Float16* __restrict__ k,
    const _Float16* __restrict__ vT, float* __restrict__ out)
{
  __shared__ __align__(16) _Float16 Plds[4][16][40];

  const int tid  = threadIdx.x;
  const int w    = tid >> 5;
  const int lane = tid & 31;
  const int lrow = lane & 15;
  const int half = lane >> 4;
  const int b = blockIdx.z, h = blockIdx.y;
  const int qbase = blockIdx.x * 64 + w * 16;

  // Q fragments (A-operand), d_k = 64 -> two K=32 chunks
  const _Float16* qrow =
      q + (((size_t)b * S_LEN + (qbase + lrow)) * NH + h) * DK;
  HV qf0, qf1;
  qf0.h[0] = *(const h8*)(qrow + half * 8);
  qf0.h[1] = *(const h8*)(qrow + 16 + half * 8);
  qf1.h[0] = *(const h8*)(qrow + 32 + half * 8);
  qf1.h[1] = *(const h8*)(qrow + 48 + half * 8);

  const _Float16* kbh = k + (size_t)b * S_LEN * D_MODEL + (size_t)h * DK;
  const _Float16* vbh = vT + ((size_t)b * NH + h) * (size_t)DK * S_LEN;

  float mstate[8], lstate[8];
  v8f oacc[4];
  for (int i = 0; i < 8; ++i) { mstate[i] = -3.0e30f; lstate[i] = 0.f; }
  for (int t = 0; t < 4; ++t) oacc[t] = zero8();

  const int nblk = CAUSAL ? ((qbase + 16 + 31) >> 5) : (S_LEN >> 5);
  for (int blk = 0; blk < nblk; ++blk) {
    const int t0 = blk * 32;
    // K^T B-fragments: column n = key index, contraction = d
    const _Float16* kr0 = kbh + (size_t)(t0 + lrow) * D_MODEL;
    const _Float16* kr1 = kr0 + (size_t)16 * D_MODEL;
    HV kf00, kf01, kf10, kf11;
    kf00.h[0] = *(const h8*)(kr0 + half * 8);
    kf00.h[1] = *(const h8*)(kr0 + 16 + half * 8);
    kf01.h[0] = *(const h8*)(kr0 + 32 + half * 8);
    kf01.h[1] = *(const h8*)(kr0 + 48 + half * 8);
    kf10.h[0] = *(const h8*)(kr1 + half * 8);
    kf10.h[1] = *(const h8*)(kr1 + 16 + half * 8);
    kf11.h[0] = *(const h8*)(kr1 + 32 + half * 8);
    kf11.h[1] = *(const h8*)(kr1 + 48 + half * 8);

    v8f s0 = wmma_f16(qf0.v, kf00.v, zero8());
    s0     = wmma_f16(qf1.v, kf01.v, s0);
    v8f s1 = wmma_f16(qf0.v, kf10.v, zero8());
    s1     = wmma_f16(qf1.v, kf11.v, s1);

    // Online softmax; each score row spans 16 lanes at one VGPR index.
    for (int i = 0; i < 8; ++i) {
      float a0 = s0[i] * 0.125f;
      float a1 = s1[i] * 0.125f;
      if (CAUSAL) {
        int gm = qbase + i + half * 8;
        if (t0 + lrow > gm)      a0 = -3.0e30f;
        if (t0 + 16 + lrow > gm) a1 = -3.0e30f;
      }
      float r = fmaxf(a0, a1);
      for (int off = 1; off < 16; off <<= 1)
        r = fmaxf(r, __shfl_xor(r, off, 32));
      float mn    = fmaxf(mstate[i], r);
      float alpha = __expf(mstate[i] - mn);
      float p0 = __expf(a0 - mn);
      float p1 = __expf(a1 - mn);
      float rs = p0 + p1;
      for (int off = 1; off < 16; off <<= 1)
        rs += __shfl_xor(rs, off, 32);
      lstate[i] = lstate[i] * alpha + rs;
      mstate[i] = mn;
      oacc[0][i] *= alpha; oacc[1][i] *= alpha;
      oacc[2][i] *= alpha; oacc[3][i] *= alpha;
      // C-layout -> LDS [m][t] so we can reload in A-fragment layout
      Plds[w][i + half * 8][lrow]      = (_Float16)p0;
      Plds[w][i + half * 8][16 + lrow] = (_Float16)p1;
    }
    asm volatile("s_wait_dscnt 0" ::: "memory");

    HV pf;
    pf.h[0] = *(const h8*)&Plds[w][lrow][half * 8];
    pf.h[1] = *(const h8*)&Plds[w][lrow][16 + half * 8];

    // P(16x32) x V(32x64): V^T fragments, column n = d, contraction = t
    const _Float16* vr = vbh + (size_t)lrow * S_LEN + t0;
    for (int dt = 0; dt < 4; ++dt) {
      HV vf;
      vf.h[0] = *(const h8*)(vr + half * 8);
      vf.h[1] = *(const h8*)(vr + 16 + half * 8);
      oacc[dt] = wmma_f16(pf.v, vf.v, oacc[dt]);
      vr += (size_t)16 * S_LEN;
    }
  }

  float* obase = out + (size_t)b * S_LEN * D_MODEL + h * DK;
  for (int i = 0; i < 8; ++i) {
    int ss = qbase + i + half * 8;
    float iv = 1.f / lstate[i];
    for (int dt = 0; dt < 4; ++dt)
      obase[(size_t)ss * D_MODEL + dt * 16 + lrow] = oacc[dt][i] * iv;
  }
}

// ---------------------------------------------------------------------------
// out = LayerNorm(x + y) * g + be, one 256-thread block per 1024-wide row.
// float4 path: each thread owns 4 contiguous columns.
// ---------------------------------------------------------------------------
__global__ __launch_bounds__(256) void ln_residual(
    const float* __restrict__ x, const float* __restrict__ y,
    const float* __restrict__ g, const float* __restrict__ be,
    float* __restrict__ out)
{
  __shared__ float rs[256], rq[256];
  const int row = blockIdx.x, tid = threadIdx.x;
  const int j0 = tid * 4;
  f4 xv = *(const f4*)(x + (size_t)row * D_MODEL + j0);
  f4 yv = *(const f4*)(y + (size_t)row * D_MODEL + j0);
  f4 v;
  float s = 0.f, s2 = 0.f;
  for (int t = 0; t < 4; ++t) {
    float a = xv[t] + yv[t];
    v[t] = a; s += a; s2 += a * a;
  }
  rs[tid] = s; rq[tid] = s2;
  __syncthreads();
  for (int off = 128; off > 0; off >>= 1) {
    if (tid < off) { rs[tid] += rs[tid + off]; rq[tid] += rq[tid + off]; }
    __syncthreads();
  }
  float mu   = rs[0] * (1.f / D_MODEL);
  float var  = rq[0] * (1.f / D_MODEL) - mu * mu;
  float rstd = rsqrtf(var + 1e-5f);
  f4 gv = *(const f4*)(g + j0);
  f4 bv = *(const f4*)(be + j0);
  f4 o;
  for (int t = 0; t < 4; ++t) o[t] = (v[t] - mu) * rstd * gv[t] + bv[t];
  *(f4*)(out + (size_t)row * D_MODEL + j0) = o;
}

// ---------------------------------------------------------------------------
extern "C" void kernel_launch(void* const* d_in, const int* in_sizes, int n_in,
                              void* d_out, int out_size, void* d_ws,
                              size_t ws_size, hipStream_t stream) {
  (void)in_sizes; (void)n_in; (void)out_size; (void)ws_size;
  const float* x    = (const float*)d_in[0];
  const float* enc  = (const float*)d_in[1];
  // d_in[2] = mask (causal handled analytically)
  const float* Wq_s = (const float*)d_in[3];
  const float* bq_s = (const float*)d_in[4];
  const float* Wk_s = (const float*)d_in[5];
  const float* bk_s = (const float*)d_in[6];
  const float* Wv_s = (const float*)d_in[7];
  const float* bv_s = (const float*)d_in[8];
  const float* Wq_x = (const float*)d_in[9];
  const float* bq_x = (const float*)d_in[10];
  const float* Wk_x = (const float*)d_in[11];
  const float* bk_x = (const float*)d_in[12];
  const float* Wv_x = (const float*)d_in[13];
  const float* bv_x = (const float*)d_in[14];
  const float* Wo   = (const float*)d_in[15];
  const float* bo   = (const float*)d_in[16];
  const float* W1   = (const float*)d_in[17];
  const float* b1   = (const float*)d_in[18];
  const float* W2   = (const float*)d_in[19];
  const float* b2   = (const float*)d_in[20];
  const float* g1   = (const float*)d_in[21];
  const float* be1  = (const float*)d_in[22];
  const float* g2   = (const float*)d_in[23];
  const float* be2  = (const float*)d_in[24];
  const float* g3   = (const float*)d_in[25];
  const float* be3  = (const float*)d_in[26];

  char* ws = (char*)d_ws;
  _Float16* qf  = (_Float16*)(ws + ((size_t)0  << 20));  //  8 MB f16 [b,s,h,d]
  _Float16* kf  = (_Float16*)(ws + ((size_t)8  << 20));  //  8 MB f16 [b,s,h,d]
  _Float16* vtf = (_Float16*)(ws + ((size_t)16 << 20));  //  8 MB f16 [b,h,d,s]
  float*    t1  = (float*)   (ws + ((size_t)24 << 20));  // 16 MB attn out
  float*    t2  = (float*)   (ws + ((size_t)40 << 20));  // 16 MB proj out
  float*    x1b = (float*)   (ws + ((size_t)56 << 20));  // 16 MB
  float*    x2b = (float*)   (ws + ((size_t)72 << 20));  // 16 MB
  float*    hb  = (float*)   (ws + ((size_t)88 << 20));  // 64 MB FFN hidden
  float*    outp = (float*)d_out;

  const int M = NB * S_LEN;                       // 4096 rows
  dim3 b256(256), b128(128);
  dim3 gP(D_MODEL / 64, M / 64);                  // 16 x 64
  dim3 gF1(FFN_DIM / 64, M / 64);                 // 64 x 64
  dim3 gA(S_LEN / 64, NH, NB);

  // ---- self-attention ----
  gemm_wmma<_Float16, false, false, true><<<gP, b256, 0, stream>>>(
      x, Wq_s, bq_s, qf, M, D_MODEL, D_MODEL);
  gemm_wmma<_Float16, false, false, true><<<gP, b256, 0, stream>>>(
      x, Wk_s, bk_s, kf, M, D_MODEL, D_MODEL);
  gemm_wmma<_Float16, false, true, true><<<gP, b256, 0, stream>>>(
      x, Wv_s, bv_s, vtf, M, D_MODEL, D_MODEL);
  attn_wmma<true><<<gA, b128, 0, stream>>>(qf, kf, vtf, t1);
  gemm_wmma<float, false, false, false><<<gP, b256, 0, stream>>>(
      t1, Wo, bo, t2, M, D_MODEL, D_MODEL);
  ln_residual<<<M, b256, 0, stream>>>(x, t2, g1, be1, x1b);

  // ---- cross-attention ----
  gemm_wmma<_Float16, false, false, true><<<gP, b256, 0, stream>>>(
      x1b, Wq_x, bq_x, qf, M, D_MODEL, D_MODEL);
  gemm_wmma<_Float16, false, false, true><<<gP, b256, 0, stream>>>(
      enc, Wk_x, bk_x, kf, M, D_MODEL, D_MODEL);
  gemm_wmma<_Float16, false, true, true><<<gP, b256, 0, stream>>>(
      enc, Wv_x, bv_x, vtf, M, D_MODEL, D_MODEL);
  attn_wmma<false><<<gA, b128, 0, stream>>>(qf, kf, vtf, t1);
  gemm_wmma<float, false, false, false><<<gP, b256, 0, stream>>>(
      t1, Wo, bo, t2, M, D_MODEL, D_MODEL);
  ln_residual<<<M, b256, 0, stream>>>(x1b, t2, g2, be2, x2b);

  // ---- FFN ----
  gemm_wmma<float, true, false, false><<<gF1, b256, 0, stream>>>(
      x2b, W1, b1, hb, M, FFN_DIM, D_MODEL);
  gemm_wmma<float, false, false, false><<<gP, b256, 0, stream>>>(
      hb, W2, b2, t2, M, D_MODEL, FFN_DIM);
  ln_residual<<<M, b256, 0, stream>>>(x2b, t2, g3, be3, outp);
}
